// LearnedSparsityG_ii_62319975465266
// MI455X (gfx1250) — matrix operand order repaired and checked by reading, not compiled, not run on hardware
//
#include <hip/hip_runtime.h>
#include <hip/hip_bf16.h>

typedef __attribute__((ext_vector_type(16))) _Float16 v16h;
typedef __attribute__((ext_vector_type(8)))  _Float16 v8h;
typedef __attribute__((ext_vector_type(4)))  _Float16 v4h;
typedef __attribute__((ext_vector_type(8)))  float    v8f;

#define DEVI __device__ __forceinline__

constexpr int B_    = 4;
constexpr int NSEQ  = 8192;
constexpr int DM    = 1024;
constexpr int HID   = 512;
constexpr int SCH   = 128;          // scan chunk length
constexpr int NCHNK = NSEQ / SCH;   // 64 chunks per (batch,dir)

// ---------------- workspace layout (bytes) ----------------
constexpr size_t OFF_OUT1   = 0;                                        // (B,N,512) f16
constexpr size_t SZ_OUT1    = (size_t)B_*NSEQ*HID*2;
constexpr size_t OFF_FEATS  = OFF_OUT1 + SZ_OUT1;                       // (B,N,512) f16
constexpr size_t OFF_LOGITS = OFF_FEATS + SZ_OUT1;                      // (B,N) f32
constexpr size_t OFF_LS     = OFF_LOGITS + (size_t)B_*NSEQ*4;           // (B,N) cpx
constexpr size_t OFF_RS     = OFF_LS + (size_t)B_*NSEQ*8;               // (B,N) cpx
constexpr size_t OFF_CHM    = OFF_RS + (size_t)B_*NSEQ*8;               // 512 mats * 8 f32
constexpr size_t OFF_ENTRY  = OFF_CHM + (size_t)B_*2*NCHNK*8*4;         // 512 cpx
constexpr size_t OFF_SPARSE = OFF_ENTRY + (size_t)B_*2*NCHNK*2*4;       // (B,N,2) f32
constexpr size_t OFF_P1     = OFF_SPARSE + (size_t)B_*NSEQ*2*4;         // (B,N,32) f32
constexpr size_t OFF_P2     = OFF_P1 + (size_t)B_*NSEQ*32*4;            // (B,N,32) f32
constexpr size_t OFF_SUM    = OFF_P2 + (size_t)B_*NSEQ*32*4;            // 1 f32

// ---------------- complex helpers ----------------
struct cpx { float re, im; };
DEVI cpx cmul(cpx a, cpx b){ return { a.re*b.re - a.im*b.im, a.re*b.im + a.im*b.re }; }
DEVI cpx cadd(cpx a, cpx b){ return { a.re+b.re, a.im+b.im }; }
DEVI cpx cscale(cpx a, float s){ return { a.re*s, a.im*s }; }
DEVI cpx crcp(cpx a){ float d = a.re*a.re + a.im*a.im; float inv = 1.0f/d; return { a.re*inv, -a.im*inv }; }
DEVI cpx cdiv(cpx a, cpx b){ return cmul(a, crcp(b)); }

// ---------------- WMMA fragment loaders ----------------
// CDNA5 wave32 16x16x32 f16 layout; LDS rows are 16B aligned (LDA/LDB = 40
// halves = 80B), so each fragment is exactly two ds_load_b128.
DEVI v16h load_a_frag(const _Float16* sA, int row, int lda, int hi) {
  const _Float16* p = sA + row*lda;
  const v8h lo = *(const v8h*)(p + (hi ? 8  : 0));
  const v8h hh = *(const v8h*)(p + (hi ? 24 : 16));
  return __builtin_shufflevector(lo, hh, 0,1,2,3,4,5,6,7,8,9,10,11,12,13,14,15);
}
DEVI v16h load_b_frag(const _Float16* sB, int col, int ldb, int hi) {
  const _Float16* p = sB + col*ldb + hi*8;
  const v8h lo = *(const v8h*)(p);
  const v8h hh = *(const v8h*)(p + 16);
  return __builtin_shufflevector(lo, hh, 0,1,2,3,4,5,6,7,8,9,10,11,12,13,14,15);
}

// ============================================================
// Conv1d(kernel=3, pad=1) as WMMA GEMM:  out[b,n,o] = relu(bias[o] +
//   sum_t sum_i in[b,n+t-1,i] * w[o,i,t]),  out stored f16 (B,N,512).
// Block tile: 64 seq rows x 128 out-channels, 8 waves, 256 threads.
// ============================================================
template<int KIN, typename TIN>
__global__ __launch_bounds__(256) void ctx_conv_kernel(const TIN* __restrict__ in,
                                                       const float* __restrict__ w,
                                                       const float* __restrict__ bias,
                                                       _Float16* __restrict__ out) {
  constexpr int LDA = 40, LDB = 40;   // 80-byte rows -> 16B-aligned b128 LDS ops
  __shared__ __align__(16) _Float16 sA[66*LDA];     // haloed A tile (rows m0-1..m0+64)
  __shared__ __align__(16) _Float16 sB[3][128*LDB]; // 3 taps of weights, K chunk of 32
  const int tid  = threadIdx.x;
  const int wave = tid >> 5;
  const int lane = tid & 31;
  const int l16  = lane & 15;
  const int hi   = lane >> 4;
  const int m0   = blockIdx.x * 64;
  const int n0   = blockIdx.y * 128;
  const int b    = blockIdx.z;

  v8f acc[4] = {};

  for (int k0 = 0; k0 < KIN; k0 += 32) {
    // ---- stage A: 66 rows x 4 octets (wide global load + ds_store_b128) ----
    for (int idx = tid; idx < 66*4; idx += 256) {
      int r = idx >> 2, c8 = (idx & 3) * 8;
      int seq = m0 + r - 1;
      v8h pk = {};
      if (seq >= 0 && seq < NSEQ) {
        const TIN* gp = &in[((size_t)b*NSEQ + seq)*KIN + (k0 + c8)];
        if constexpr (sizeof(TIN) == 4) {
          v8f g = *(const v8f*)gp;
#pragma unroll
          for (int j = 0; j < 8; ++j) pk[j] = (_Float16)g[j];
        } else {
          pk = *(const v8h*)gp;
        }
      }
      *(v8h*)&sA[r*LDA + c8] = pk;
    }
    // ---- stage B: 3 taps x 128 outputs x 4 octets (strided gather, b128 store) ----
    for (int idx = tid; idx < 3*128*4; idx += 256) {
      int t = idx >> 9;             // /512
      int rem = idx & 511;
      int o = rem >> 2, c8 = (rem & 3) * 8;
      const float* gp = &w[((size_t)(n0 + o)*KIN + (k0 + c8))*3 + t];
      v8h pk;
#pragma unroll
      for (int j = 0; j < 8; ++j) pk[j] = (_Float16)gp[3*j];
      *(v8h*)&sB[t][o*LDB + c8] = pk;
    }
    if (k0 + 32 < KIN)
      __builtin_prefetch(&in[((size_t)b*NSEQ + m0)*KIN + (k0 + 32)], 0, 1);
    __syncthreads();

#pragma unroll
    for (int t = 0; t < 3; ++t) {
      v16h bf = load_b_frag(sB[t], wave*16 + l16, LDB, hi);
#pragma unroll
      for (int ms = 0; ms < 4; ++ms) {
        v16h af = load_a_frag(sA, ms*16 + l16 + t, LDA, hi);
        acc[ms] = __builtin_amdgcn_wmma_f32_16x16x32_f16(
            false, af, false, bf, (short)0, acc[ms], false, false);
      }
    }
    __syncthreads();
  }

  const int ocol = n0 + wave*16 + l16;
  const float bv = bias[ocol];
#pragma unroll
  for (int ms = 0; ms < 4; ++ms) {
#pragma unroll
    for (int r = 0; r < 8; ++r) {
      int m = m0 + ms*16 + hi*8 + r;
      float vo = acc[ms][r] + bv;
      vo = vo > 0.0f ? vo : 0.0f;
      out[((size_t)b*NSEQ + m)*HID + ocol] = (_Float16)vo;
    }
  }
}

// ============================================================
// Fused: hh = relu(feats @ w_imp1^T + b1); logits = hh @ w_imp2^T + b2
// Block: 64 seq rows, loops over 4 o-chunks of 128; WMMA accum f32.
// ============================================================
__global__ __launch_bounds__(256) void imp_kernel(const _Float16* __restrict__ feats,
                                                  const float* __restrict__ w1,
                                                  const float* __restrict__ b1,
                                                  const float* __restrict__ w2,
                                                  const float* __restrict__ b2,
                                                  float* __restrict__ logits) {
  constexpr int LDA = 40, LDB = 40;
  __shared__ __align__(16) _Float16 sA[64*LDA];
  __shared__ __align__(16) _Float16 sB[128*LDB];
  __shared__ float logitLds[64];
  const int tid  = threadIdx.x;
  const int wave = tid >> 5;
  const int lane = tid & 31;
  const int l16  = lane & 15;
  const int hi   = lane >> 4;
  const int m0   = blockIdx.x * 64;
  const int b    = blockIdx.z;

  if (tid < 64) logitLds[tid] = 0.0f;
  __syncthreads();

  float lg[32] = {};   // per-lane partial logits: [ms*8 + r]

  for (int oc = 0; oc < 4; ++oc) {
    const int obase = oc * 128;
    v8f acc[4] = {};
    for (int k0 = 0; k0 < HID; k0 += 32) {
      for (int idx = tid; idx < 64*4; idx += 256) {
        int r = idx >> 2, c8 = (idx & 3) * 8;
        *(v8h*)&sA[r*LDA + c8] =
            *(const v8h*)&feats[((size_t)b*NSEQ + m0 + r)*HID + (k0 + c8)];
      }
      for (int idx = tid; idx < 128*4; idx += 256) {
        int o = idx >> 2, c8 = (idx & 3) * 8;
        v8f g = *(const v8f*)&w1[(size_t)(obase + o)*HID + (k0 + c8)];
        v8h pk;
#pragma unroll
        for (int j = 0; j < 8; ++j) pk[j] = (_Float16)g[j];
        *(v8h*)&sB[o*LDB + c8] = pk;
      }
      __syncthreads();
      v16h bf = load_b_frag(sB, wave*16 + l16, LDB, hi);
#pragma unroll
      for (int ms = 0; ms < 4; ++ms) {
        v16h af = load_a_frag(sA, ms*16 + l16, LDA, hi);
        acc[ms] = __builtin_amdgcn_wmma_f32_16x16x32_f16(
            false, af, false, bf, (short)0, acc[ms], false, false);
      }
      __syncthreads();
    }
    const int o = obase + wave*16 + l16;
    const float bb = b1[o], ww = w2[o];
#pragma unroll
    for (int ms = 0; ms < 4; ++ms) {
#pragma unroll
      for (int r = 0; r < 8; ++r) {
        float h = acc[ms][r] + bb;
        h = h > 0.0f ? h : 0.0f;
        lg[ms*8 + r] += h * ww;
      }
    }
  }

  // reduce across the 16 lanes of each half-wave (same row group)
#pragma unroll
  for (int i = 0; i < 32; ++i) {
    float t = lg[i];
    t += __shfl_xor(t, 1);
    t += __shfl_xor(t, 2);
    t += __shfl_xor(t, 4);
    t += __shfl_xor(t, 8);
    lg[i] = t;
  }
  if (l16 == 0) {
#pragma unroll
    for (int ms = 0; ms < 4; ++ms)
#pragma unroll
      for (int r = 0; r < 8; ++r)
        atomicAdd(&logitLds[ms*16 + hi*8 + r], lg[ms*8 + r]);
  }
  __syncthreads();
  if (tid < 64)
    logits[(size_t)b*NSEQ + m0 + tid] = logitLds[tid] + b2[0];
}

// ============================================================
// Tridiagonal Green's function scan (c == 1):  L_i = (i - d_i) - c_i/L_{i-1}
// as Mobius matrices M_i = [[z-d_i, -c_i],[1,0]];  3-phase chunked scan.
// ============================================================
__global__ void scan_chunk_kernel(const float* __restrict__ v, float* __restrict__ chunkM) {
  int gid = blockIdx.x * blockDim.x + threadIdx.x;
  if (gid >= B_*2*NCHNK) return;
  int chunk = gid % NCHNK;
  int bd = gid / NCHNK;
  int dir = bd & 1;
  int b = bd >> 1;
  cpx p00 = {1,0}, p01 = {0,0}, p10 = {0,0}, p11 = {1,0};
  for (int s = 0; s < SCH; ++s) {
    int pos = chunk*SCH + s;
    int i = dir ? (NSEQ - 1 - pos) : pos;
    float d = v[(size_t)b*NSEQ + i] - 2.0f;
    cpx a = { -d, 1.0f };
    float cf = (pos == 0) ? 0.0f : 1.0f;
    cpx n00 = { a.re*p00.re - a.im*p00.im - cf*p10.re, a.re*p00.im + a.im*p00.re - cf*p10.im };
    cpx n01 = { a.re*p01.re - a.im*p01.im - cf*p11.re, a.re*p01.im + a.im*p01.re - cf*p11.im };
    p10 = p00; p11 = p01; p00 = n00; p01 = n01;
    float mag = fabsf(p00.re) + fabsf(p00.im) + fabsf(p10.re) + fabsf(p10.im);
    float sc = 1.0f / (mag + 1e-30f);
    p00 = cscale(p00, sc); p01 = cscale(p01, sc);
    p10 = cscale(p10, sc); p11 = cscale(p11, sc);
  }
  float* dst = chunkM + (size_t)gid*8;
  dst[0]=p00.re; dst[1]=p00.im; dst[2]=p01.re; dst[3]=p01.im;
  dst[4]=p10.re; dst[5]=p10.im; dst[6]=p11.re; dst[7]=p11.im;
}

__global__ void scan_prefix_kernel(const float* __restrict__ chunkM, float* __restrict__ entryL) {
  int t = threadIdx.x;
  if (t >= B_*2) return;
  cpx q00 = {1,0}, q01 = {0,0}, q10 = {0,0}, q11 = {1,0};
  for (int c = 0; c < NCHNK; ++c) {
    cpx e = cdiv(cadd(q00, q01), cadd(q10, q11));   // apply prefix to L=1
    float* ep = entryL + ((size_t)t*NCHNK + c)*2;
    ep[0] = e.re; ep[1] = e.im;
    const float* mp = chunkM + ((size_t)t*NCHNK + c)*8;
    cpx m00 = {mp[0],mp[1]}, m01 = {mp[2],mp[3]}, m10 = {mp[4],mp[5]}, m11 = {mp[6],mp[7]};
    cpx r00 = cadd(cmul(m00,q00), cmul(m01,q10));
    cpx r01 = cadd(cmul(m00,q01), cmul(m01,q11));
    cpx r10 = cadd(cmul(m10,q00), cmul(m11,q10));
    cpx r11 = cadd(cmul(m10,q01), cmul(m11,q11));
    float mag = fabsf(r00.re) + fabsf(r00.im) + fabsf(r10.re) + fabsf(r10.im);
    float sc = 1.0f / (mag + 1e-30f);
    q00 = cscale(r00, sc); q01 = cscale(r01, sc);
    q10 = cscale(r10, sc); q11 = cscale(r11, sc);
  }
}

__global__ void scan_local_kernel(const float* __restrict__ v, const float* __restrict__ entryL,
                                  float* __restrict__ Ls, float* __restrict__ Rs) {
  int gid = blockIdx.x * blockDim.x + threadIdx.x;
  if (gid >= B_*2*NCHNK) return;
  int chunk = gid % NCHNK;
  int bd = gid / NCHNK;
  int dir = bd & 1;
  int b = bd >> 1;
  const float* ep = entryL + (size_t)gid*2;
  cpx L = { ep[0], ep[1] };
  float* dstBase = (dir ? Rs : Ls);
  for (int s = 0; s < SCH; ++s) {
    int pos = chunk*SCH + s;
    int i = dir ? (NSEQ - 1 - pos) : pos;
    float d = v[(size_t)b*NSEQ + i] - 2.0f;
    float cf = (pos == 0) ? 0.0f : 1.0f;
    cpx inv = crcp(L);
    L = { -d - cf*inv.re, 1.0f - cf*inv.im };
    float* dst = dstBase + ((size_t)b*NSEQ + i)*2;
    dst[0] = L.re; dst[1] = L.im;
  }
}

// ============================================================
// G = 1/(L + R - (z-d)); mask = (logits + gumbel > 0); sparse = G*mask
// ============================================================
__global__ __launch_bounds__(256) void green_mask_kernel(const float* __restrict__ v,
                                                         const float* __restrict__ u,
                                                         const float* __restrict__ logits,
                                                         const float* __restrict__ Ls,
                                                         const float* __restrict__ Rs,
                                                         float* __restrict__ sparse,
                                                         float* __restrict__ maskOut,
                                                         float* __restrict__ sumPtr) {
  __shared__ float red[256];
  int idx = blockIdx.x * blockDim.x + threadIdx.x;
  float mk = 0.0f;
  if (idx < B_*NSEQ) {
    float d = v[idx] - 2.0f;
    cpx a = { -d, 1.0f };
    cpx L = { Ls[2*idx], Ls[2*idx+1] };
    cpx R = { Rs[2*idx], Rs[2*idx+1] };
    cpx den = { L.re + R.re - a.re, L.im + R.im - a.im };
    cpx G = crcp(den);
    float g = -logf(-logf(u[idx] + 1e-10f) + 1e-10f);
    mk = (logits[idx] + g > 0.0f) ? 1.0f : 0.0f;
    maskOut[idx] = mk;
    sparse[2*idx]   = G.re * mk;
    sparse[2*idx+1] = G.im * mk;
  }
  red[threadIdx.x] = mk;
  __syncthreads();
  for (int s = 128; s > 0; s >>= 1) {
    if (threadIdx.x < s) red[threadIdx.x] += red[threadIdx.x + s];
    __syncthreads();
  }
  if (threadIdx.x == 0) atomicAdd(sumPtr, red[0]);
}

// ============================================================
// Small interpolation convs (kernel=5, pad=2)
// ============================================================
__global__ __launch_bounds__(256) void int_conv1_kernel(const float* __restrict__ sparse,
                                                        const float* __restrict__ w,
                                                        const float* __restrict__ bias,
                                                        float* __restrict__ p1) {
  __shared__ float sW[32*2*5];
  for (int i = threadIdx.x; i < 32*2*5; i += 256) sW[i] = w[i];
  __syncthreads();
  int idx = blockIdx.x * blockDim.x + threadIdx.x;
  if (idx >= B_*NSEQ) return;
  int b = idx / NSEQ, n = idx % NSEQ;
  float in[5][2];
#pragma unroll
  for (int t = 0; t < 5; ++t) {
    int m = n + t - 2;
    bool ok = (m >= 0 && m < NSEQ);
    in[t][0] = ok ? sparse[((size_t)b*NSEQ + m)*2 + 0] : 0.0f;
    in[t][1] = ok ? sparse[((size_t)b*NSEQ + m)*2 + 1] : 0.0f;
  }
  for (int o = 0; o < 32; ++o) {
    float s = bias[o];
#pragma unroll
    for (int t = 0; t < 5; ++t) {
      s += in[t][0] * sW[(o*2 + 0)*5 + t];
      s += in[t][1] * sW[(o*2 + 1)*5 + t];
    }
    p1[((size_t)b*NSEQ + n)*32 + o] = s > 0.0f ? s : 0.0f;
  }
}

__global__ __launch_bounds__(256) void int_conv2_kernel(const float* __restrict__ p1,
                                                        const float* __restrict__ w,
                                                        const float* __restrict__ bias,
                                                        float* __restrict__ p2) {
  __shared__ float sW[32*32*5];
  for (int i = threadIdx.x; i < 32*32*5; i += 256) sW[i] = w[i];
  __syncthreads();
  int idx = blockIdx.x * blockDim.x + threadIdx.x;
  if (idx >= B_*NSEQ) return;
  int b = idx / NSEQ, n = idx % NSEQ;
  float acc[32];
#pragma unroll
  for (int o = 0; o < 32; ++o) acc[o] = bias[o];
  for (int t = 0; t < 5; ++t) {
    int m = n + t - 2;
    if (m < 0 || m >= NSEQ) continue;
    const float* pin = p1 + ((size_t)b*NSEQ + m)*32;
    for (int i = 0; i < 32; ++i) {
      float xv = pin[i];
#pragma unroll
      for (int o = 0; o < 32; ++o) acc[o] += xv * sW[(o*32 + i)*5 + t];
    }
  }
#pragma unroll
  for (int o = 0; o < 32; ++o)
    p2[((size_t)b*NSEQ + n)*32 + o] = acc[o] > 0.0f ? acc[o] : 0.0f;
}

__global__ __launch_bounds__(256) void int_conv3_final_kernel(const float* __restrict__ p2,
                                                              const float* __restrict__ w,
                                                              const float* __restrict__ bias,
                                                              const float* __restrict__ sparse,
                                                              const float* __restrict__ maskOut,
                                                              float* __restrict__ finalOut) {
  __shared__ float sW[2*32*5];
  for (int i = threadIdx.x; i < 2*32*5; i += 256) sW[i] = w[i];
  __syncthreads();
  int idx = blockIdx.x * blockDim.x + threadIdx.x;
  if (idx >= B_*NSEQ) return;
  int b = idx / NSEQ, n = idx % NSEQ;
  float acc0 = bias[0], acc1 = bias[1];
  for (int t = 0; t < 5; ++t) {
    int m = n + t - 2;
    if (m < 0 || m >= NSEQ) continue;
    const float* pin = p2 + ((size_t)b*NSEQ + m)*32;
    for (int i = 0; i < 32; ++i) {
      float xv = pin[i];
      acc0 += xv * sW[(0*32 + i)*5 + t];
      acc1 += xv * sW[(1*32 + i)*5 + t];
    }
  }
  float mk = maskOut[idx];
  float s0 = sparse[2*idx], s1 = sparse[2*idx+1];
  finalOut[2*idx]   = (mk > 0.5f) ? s0 : acc0;
  finalOut[2*idx+1] = (mk > 0.5f) ? s1 : acc1;
}

__global__ void init_sum_kernel(float* sumPtr) { *sumPtr = 0.0f; }

__global__ void finalize_kernel(const float* sumPtr, float* ratioOut) {
  *ratioOut = 1.0f - (*sumPtr) / (float)(B_*NSEQ);
}

// ============================================================
extern "C" void kernel_launch(void* const* d_in, const int* in_sizes, int n_in,
                              void* d_out, int out_size, void* d_ws, size_t ws_size,
                              hipStream_t stream) {
  const float* x      = (const float*)d_in[0];
  const float* v      = (const float*)d_in[1];
  const float* u      = (const float*)d_in[2];
  const float* w_ctx1 = (const float*)d_in[3];
  const float* b_ctx1 = (const float*)d_in[4];
  const float* w_ctx2 = (const float*)d_in[5];
  const float* b_ctx2 = (const float*)d_in[6];
  const float* w_imp1 = (const float*)d_in[7];
  const float* b_imp1 = (const float*)d_in[8];
  const float* w_imp2 = (const float*)d_in[9];
  const float* b_imp2 = (const float*)d_in[10];
  const float* w_int1 = (const float*)d_in[11];
  const float* b_int1 = (const float*)d_in[12];
  const float* w_int2 = (const float*)d_in[13];
  const float* b_int2 = (const float*)d_in[14];
  const float* w_int3 = (const float*)d_in[15];
  const float* b_int3 = (const float*)d_in[16];

  float* out = (float*)d_out;
  float* finalOut = out;                          // (B,N,2)
  float* maskOut  = out + (size_t)B_*NSEQ*2;      // (B,N)
  float* ratioOut = out + (size_t)B_*NSEQ*3;      // scalar

  char* ws = (char*)d_ws;
  _Float16* out1   = (_Float16*)(ws + OFF_OUT1);
  _Float16* feats  = (_Float16*)(ws + OFF_FEATS);
  float*    logits = (float*)(ws + OFF_LOGITS);
  float*    Ls     = (float*)(ws + OFF_LS);
  float*    Rs     = (float*)(ws + OFF_RS);
  float*    chunkM = (float*)(ws + OFF_CHM);
  float*    entryL = (float*)(ws + OFF_ENTRY);
  float*    sparse = (float*)(ws + OFF_SPARSE);
  float*    p1     = (float*)(ws + OFF_P1);
  float*    p2     = (float*)(ws + OFF_P2);
  float*    sumPtr = (float*)(ws + OFF_SUM);

  const dim3 gGemm(NSEQ/64, HID/128, B_);
  const int  nElemBlocks = (B_*NSEQ + 255) / 256;

  init_sum_kernel<<<1, 1, 0, stream>>>(sumPtr);

  // context convs as WMMA GEMMs
  ctx_conv_kernel<DM,  float>   <<<gGemm, 256, 0, stream>>>(x,    w_ctx1, b_ctx1, out1);
  ctx_conv_kernel<HID, _Float16><<<gGemm, 256, 0, stream>>>(out1, w_ctx2, b_ctx2, feats);

  // fused importance MLP -> logits
  imp_kernel<<<dim3(NSEQ/64, 1, B_), 256, 0, stream>>>(feats, w_imp1, b_imp1, w_imp2, b_imp2, logits);

  // chunked parallel Mobius scans (forward + backward)
  scan_chunk_kernel <<<(B_*2*NCHNK + 255)/256, 256, 0, stream>>>(v, chunkM);
  scan_prefix_kernel<<<1, 32, 0, stream>>>(chunkM, entryL);
  scan_local_kernel <<<(B_*2*NCHNK + 255)/256, 256, 0, stream>>>(v, entryL, Ls, Rs);

  // Green's function diagonal + Gumbel mask + sparse features + mask sum
  green_mask_kernel<<<nElemBlocks, 256, 0, stream>>>(v, u, logits, Ls, Rs, sparse, maskOut, sumPtr);

  // interpolation convs + final select
  int_conv1_kernel<<<nElemBlocks, 256, 0, stream>>>(sparse, w_int1, b_int1, p1);
  int_conv2_kernel<<<nElemBlocks, 256, 0, stream>>>(p1, w_int2, b_int2, p2);
  int_conv3_final_kernel<<<nElemBlocks, 256, 0, stream>>>(p2, w_int3, b_int3, sparse, maskOut, finalOut);

  finalize_kernel<<<1, 1, 0, stream>>>(sumPtr, ratioOut);
}
